// Head_49778670961195
// MI455X (gfx1250) — compile-verified, matrix operand order checked by reading, not run on hardware
//
#include <hip/hip_runtime.h>

#define B_ 8
#define T_ 2048
#define C_ 1024
#define D_ 128

typedef __attribute__((ext_vector_type(16))) __bf16   v16bf;
typedef __attribute__((ext_vector_type(2)))  __bf16   v2bf;
typedef __attribute__((ext_vector_type(8)))  float    v8f;
typedef __attribute__((ext_vector_type(8)))  unsigned v8u;

// ---------- helpers ----------
// Let the hardware do fp32->bf16 (RNE fptrunc -> v_cvt*_bf16_f32 family)
static __device__ __forceinline__ unsigned short f2bfu(float f) {
    __bf16 h = (__bf16)f;
    return __builtin_bit_cast(unsigned short, h);
}
static __device__ __forceinline__ unsigned pack2(float a, float b) {
    v2bf v;
    v[0] = (__bf16)a;
    v[1] = (__bf16)b;
    return __builtin_bit_cast(unsigned, v);
}
static __device__ __forceinline__ v16bf mkfrag(uint4 lo, uint4 hi) {
    v8u u;
    u[0] = lo.x; u[1] = lo.y; u[2] = lo.z; u[3] = lo.w;
    u[4] = hi.x; u[5] = hi.y; u[6] = hi.z; u[7] = hi.w;
    return __builtin_bit_cast(v16bf, u);
}

// ---------- kernel 1: W [C,D] f32 -> Wt bf16 [3][D][C] ----------
__global__ __launch_bounds__(256) void wt_kernel(const float* __restrict__ Wq,
                                                 const float* __restrict__ Wk,
                                                 const float* __restrict__ Wv,
                                                 unsigned short* __restrict__ wt) {
    int idx = blockIdx.x * 256 + threadIdx.x;     // 3*D*C total
    if (idx >= 3 * D_ * C_) return;
    int mat = idx / (D_ * C_);
    int rem = idx - mat * (D_ * C_);
    int n = rem / C_;
    int k = rem - n * C_;
    const float* W = (mat == 0) ? Wq : (mat == 1) ? Wk : Wv;
    wt[idx] = f2bfu(W[(size_t)k * D_ + n]);
}

// ---------- kernel 2: fused QKV projection ----------
// grid.x = (B*T)/16 = 1024 M-tiles, grid.y = matrix (0=Q,1=K,2=V)
// 8 waves per block; wave w owns N-tile w (D = 8 * 16).
__global__ __launch_bounds__(256) void qkv_kernel(const float* __restrict__ x,
                                                  const unsigned short* __restrict__ wt,
                                                  unsigned short* __restrict__ qb,
                                                  unsigned short* __restrict__ kb,
                                                  unsigned short* __restrict__ vtb) {
    const int wave = threadIdx.x >> 5;
    const int lane = threadIdx.x & 31;
    const int l16  = lane & 15;
    const int hsel = lane >> 4;            // which half-wave
    const int mtile = blockIdx.x;          // 0..1023
    const int mat   = blockIdx.y;          // 0..2
    const int ntile = wave;                // 0..7

    const int arow = mtile * 16 + l16;     // A-matrix row for this lane
    const float*          xrow = x  + (size_t)arow * C_;
    const unsigned short* wrow = wt + (size_t)mat * D_ * C_ + (size_t)(ntile * 16 + l16) * C_;
    const uint4* wrow4 = (const uint4*)wrow;

    v8f acc = {};
    #pragma unroll 2
    for (int kk = 0; kk < C_; kk += 32) {
        const int k0 = kk + hsel * 8;
        // A fragment: convert fp32 x on the fly (hardware bf16 converts)
        float4 f0 = *(const float4*)(xrow + k0);
        float4 f1 = *(const float4*)(xrow + k0 + 4);
        float4 f2 = *(const float4*)(xrow + k0 + 16);
        float4 f3 = *(const float4*)(xrow + k0 + 20);
        v8u au;
        au[0] = pack2(f0.x, f0.y); au[1] = pack2(f0.z, f0.w);
        au[2] = pack2(f1.x, f1.y); au[3] = pack2(f1.z, f1.w);
        au[4] = pack2(f2.x, f2.y); au[5] = pack2(f2.z, f2.w);
        au[6] = pack2(f3.x, f3.y); au[7] = pack2(f3.z, f3.w);
        v16bf a = __builtin_bit_cast(v16bf, au);
        // B fragment: contiguous bf16 from transposed weights
        v16bf b = mkfrag(wrow4[k0 >> 3], wrow4[(k0 >> 3) + 2]);
        acc = __builtin_amdgcn_wmma_f32_16x16x32_bf16(false, a, false, b,
                                                      (short)0, acc, false, false);
    }

    const int mrow0 = mtile * 16 + hsel * 8;   // C-layout: rows r (+8 for upper half)
    const int coln  = ntile * 16 + l16;
    const float qscale = 0.08838834764831845f; // D^-0.5 folded into Q
    if (mat == 0) {
        #pragma unroll
        for (int r = 0; r < 8; ++r)
            qb[(size_t)(mrow0 + r) * D_ + coln] = f2bfu(acc[r] * qscale);
    } else if (mat == 1) {
        #pragma unroll
        for (int r = 0; r < 8; ++r)
            kb[(size_t)(mrow0 + r) * D_ + coln] = f2bfu(acc[r]);
    } else {
        #pragma unroll
        for (int r = 0; r < 8; ++r) {
            int grow = mrow0 + r;
            int bb = grow >> 11;               // / T
            int t  = grow & (T_ - 1);
            vtb[(size_t)bb * D_ * T_ + (size_t)coln * T_ + t] = f2bfu(acc[r]);
        }
    }
}

// ---------- kernel 3: flash attention, one 16-query tile per wave ----------
__global__ __launch_bounds__(256) void attn_kernel(const unsigned short* __restrict__ qb,
                                                   const unsigned short* __restrict__ kb,
                                                   const unsigned short* __restrict__ vtb,
                                                   float* __restrict__ out) {
    __shared__ alignas(16) unsigned short pl[8][16 * 32];  // per-wave P transpose buffer

    const int wave = threadIdx.x >> 5;
    const int lane = threadIdx.x & 31;
    const int l16  = lane & 15;
    const int hsel = lane >> 4;

    const int qtile = blockIdx.x * 8 + wave;   // 0..1023
    const int b     = qtile >> 7;              // T/16 = 128 tiles per batch
    const int qBase = (qtile & 127) * 16;

    // Q fragments: 16x128 tile -> 4 A-fragments (K=32 each)
    const uint4* qrow = (const uint4*)(qb + (size_t)(b * T_ + qBase + l16) * D_);
    v16bf qf[4];
    #pragma unroll
    for (int f = 0; f < 4; ++f)
        qf[f] = mkfrag(qrow[f * 4 + hsel], qrow[f * 4 + hsel + 2]);

    v8f   o[8];
    float mi[8], li[8];
    #pragma unroll
    for (int t = 0; t < 8; ++t) o[t] = (v8f){};
    #pragma unroll
    for (int r = 0; r < 8; ++r) { mi[r] = -1e30f; li[r] = 0.0f; }

    const int jEnd = (qBase + 15) >> 5;        // inclusive; causal skip of upper triangle
    for (int j = 0; j <= jEnd; ++j) {
        const int kBase = j * 32;
        // S = Q K^T for 32 keys -> two 16x16 f32 tiles
        v8f s0 = {}, s1 = {};
        const uint4* kr0 = (const uint4*)(kb + (size_t)(b * T_ + kBase + l16) * D_);
        const uint4* kr1 = (const uint4*)(kb + (size_t)(b * T_ + kBase + 16 + l16) * D_);
        #pragma unroll
        for (int f = 0; f < 4; ++f) {
            v16bf kf0 = mkfrag(kr0[f * 4 + hsel], kr0[f * 4 + hsel + 2]);
            s0 = __builtin_amdgcn_wmma_f32_16x16x32_bf16(false, qf[f], false, kf0,
                                                         (short)0, s0, false, false);
            v16bf kf1 = mkfrag(kr1[f * 4 + hsel], kr1[f * 4 + hsel + 2]);
            s1 = __builtin_amdgcn_wmma_f32_16x16x32_bf16(false, qf[f], false, kf1,
                                                         (short)0, s1, false, false);
        }

        // causal mask + online softmax (row r lives in VGPR slot r, +8 for upper half)
        float alpha[8];
        #pragma unroll
        for (int r = 0; r < 8; ++r) {
            const int qi = qBase + r + 8 * hsel;
            if (kBase + l16 > qi)      s0[r] = -1e30f;
            if (kBase + 16 + l16 > qi) s1[r] = -1e30f;

            float mx = fmaxf(s0[r], s1[r]);
            #pragma unroll
            for (int off = 8; off > 0; off >>= 1)
                mx = fmaxf(mx, __shfl_xor(mx, off, 16));
            const float mn = fmaxf(mi[r], mx);
            alpha[r] = __expf(mi[r] - mn);
            mi[r] = mn;
            const float p0 = __expf(s0[r] - mn);
            const float p1 = __expf(s1[r] - mn);
            s0[r] = p0; s1[r] = p1;
            float sum = p0 + p1;
            #pragma unroll
            for (int off = 8; off > 0; off >>= 1)
                sum += __shfl_xor(sum, off, 16);
            li[r] = li[r] * alpha[r] + sum;
        }
        #pragma unroll
        for (int t = 0; t < 8; ++t)
            #pragma unroll
            for (int r = 0; r < 8; ++r)
                o[t][r] *= alpha[r];

        // transpose P (C-layout -> A-layout) through LDS, as bf16
        unsigned short* pp = pl[wave];
        #pragma unroll
        for (int r = 0; r < 8; ++r) {
            const int m = r + 8 * hsel;
            pp[m * 32 + l16]      = f2bfu(s0[r]);
            pp[m * 32 + 16 + l16] = f2bfu(s1[r]);
        }
        const uint4* prow = (const uint4*)(pp + l16 * 32);
        v16bf pf = mkfrag(prow[hsel], prow[hsel + 2]);

        // O += P * V  (V stored transposed [D,T] so B-fragments are contiguous)
        #pragma unroll
        for (int t = 0; t < 8; ++t) {
            const uint4* vr = (const uint4*)(vtb + (size_t)b * D_ * T_ +
                                             (size_t)(t * 16 + l16) * T_ + kBase);
            v16bf vf = mkfrag(vr[hsel], vr[hsel + 2]);
            o[t] = __builtin_amdgcn_wmma_f32_16x16x32_bf16(false, pf, false, vf,
                                                           (short)0, o[t], false, false);
        }
    }

    // epilogue: normalize and store fp32
    float inv[8];
    #pragma unroll
    for (int r = 0; r < 8; ++r) inv[r] = 1.0f / li[r];
    #pragma unroll
    for (int t = 0; t < 8; ++t)
        #pragma unroll
        for (int r = 0; r < 8; ++r)
            out[(size_t)(b * T_ + qBase + r + 8 * hsel) * D_ + t * 16 + l16] = o[t][r] * inv[r];
}

// ---------- launch ----------
extern "C" void kernel_launch(void* const* d_in, const int* in_sizes, int n_in,
                              void* d_out, int out_size, void* d_ws, size_t ws_size,
                              hipStream_t stream) {
    const float* x  = (const float*)d_in[0];
    const float* Wq = (const float*)d_in[1];
    const float* Wk = (const float*)d_in[2];
    const float* Wv = (const float*)d_in[3];
    float* out = (float*)d_out;

    // workspace layout (bf16 = ushort): Q | K | V^T | Wt   (~12.75 MiB total)
    unsigned short* qb  = (unsigned short*)d_ws;
    unsigned short* kb  = qb  + (size_t)B_ * T_ * D_;
    unsigned short* vtb = kb  + (size_t)B_ * T_ * D_;
    unsigned short* wt  = vtb + (size_t)B_ * T_ * D_;

    // 1) convert+transpose weights to bf16 [3][D][C]
    wt_kernel<<<(3 * D_ * C_ + 255) / 256, 256, 0, stream>>>(Wq, Wk, Wv, wt);

    // 2) fused QKV projection (Q pre-scaled by D^-0.5, V stored transposed)
    dim3 g2((B_ * T_) / 16, 3, 1);
    qkv_kernel<<<g2, 256, 0, stream>>>(x, wt, qb, kb, vtb);

    // 3) flash attention
    attn_kernel<<<(B_ * T_) / 16 / 8, 256, 0, stream>>>(qb, kb, vtb, out);
}
